// SoftDecisionTree_14809047237092
// MI455X (gfx1250) — matrix operand order repaired and checked by reading, not compiled, not run on hardware
//
#include <hip/hip_runtime.h>
#include <hip/hip_bf16.h>

typedef __attribute__((ext_vector_type(16))) __bf16 v16bf;
typedef __attribute__((ext_vector_type(8)))  float  v8f;

#define DIN   2048
#define NPAD  256
#define NBATCH 8192
#define NOUT  100
#define NNODE 255
#define NPEN  127
#define LMBDA 0.1f

// ---------------------------------------------------------------------------
// Kernel 1: transpose+convert W -> Wt[k][n] (bf16, node 255 padded to 0),
// pad b/beta to 256 entries, zero the 513-float accumulator block.
// ---------------------------------------------------------------------------
__global__ void sdt_prep(const float* __restrict__ W,
                         const float* __restrict__ b,
                         const float* __restrict__ beta,
                         __bf16* __restrict__ Wt,
                         float* __restrict__ bpad,
                         float* __restrict__ betapad,
                         float* __restrict__ accums) {
  int idx = blockIdx.x * blockDim.x + threadIdx.x;   // 0 .. DIN*NPAD-1
  int k = idx >> 8;
  int n = idx & 255;
  float w = (n < NNODE) ? W[(size_t)n * DIN + k] : 0.0f;
  Wt[idx] = (__bf16)w;                                // Wt[k*NPAD + n]
  if (idx < NPAD) {
    bpad[idx]    = (idx < NNODE) ? b[idx]    : 0.0f;
    betapad[idx] = (idx < NNODE) ? beta[idx] : 0.0f;
  }
  if (idx < 513) accums[idx] = 0.0f;                  // loss + 2*256 alpha sums
}

// ---------------------------------------------------------------------------
// Kernel 2: logQ = log_softmax(leaf_params), Q = exp(logQ). 256 rows x 100.
// ---------------------------------------------------------------------------
__global__ void sdt_softmax(const float* __restrict__ leafp,
                            float* __restrict__ logQ,
                            float* __restrict__ Q) {
  int r = threadIdx.x;                                // one thread per row
  if (r >= NPAD) return;
  const float* row = leafp + (size_t)r * NOUT;
  float mx = -INFINITY;
  for (int j = 0; j < NOUT; ++j) mx = fmaxf(mx, row[j]);
  float s = 0.0f;
  for (int j = 0; j < NOUT; ++j) s += __expf(row[j] - mx);
  float lse = mx + __logf(s);
  for (int j = 0; j < NOUT; ++j) {
    float lq = row[j] - lse;
    logQ[(size_t)r * NOUT + j] = lq;
    Q[(size_t)r * NOUT + j]    = __expf(lq);
  }
}

// ---------------------------------------------------------------------------
// Kernel 3: p = sigmoid(beta * (x @ W^T + b)) via bf16 WMMA, f32 accumulate.
// One wave computes a 16-row x 64-col tile (4 x 16x16 accumulators).
// Block = 8 waves = 128 rows. Grid = (8192/128, 256/64) = (64, 4).
// A fragment layout (16-bit A 16x32): lane m=lane&15 holds row m; lanes 0-15
// carry K offsets {0..7,16..23}, lanes 16-31 carry {8..15,24..31}.
// B fragment (32x16 bf16): lane l holds column N=l&15 with the same K offsets
// as A: 16 bf16 read down a Wt column, stride NPAD.
// ---------------------------------------------------------------------------
__global__ void sdt_gemm(const float* __restrict__ x,
                         const __bf16* __restrict__ Wt,
                         const float* __restrict__ bpad,
                         const float* __restrict__ betapad,
                         float* __restrict__ pbuf) {
  const int lane = threadIdx.x & 31;
  const int wave = threadIdx.x >> 5;
  const int row0 = (blockIdx.x * 8 + wave) * 16;
  const int col0 = blockIdx.y * 64;
  const int m    = lane & 15;
  const int half = lane >> 4;         // 0 or 1

  const float* xr = x + (size_t)(row0 + m) * DIN + half * 8;
  const int krowbase = m + half * 16; // unused; each B lane holds column m

  v8f acc[4] = {};

  for (int k0 = 0; k0 < DIN; k0 += 32) {
    // ---- A fragment: load 16 f32 of row m, convert to bf16 in-register
    float4 f0 = *(const float4*)(xr + k0);
    float4 f1 = *(const float4*)(xr + k0 + 4);
    float4 f2 = *(const float4*)(xr + k0 + 16);
    float4 f3 = *(const float4*)(xr + k0 + 20);
    v16bf a;
    a[0]  = (__bf16)f0.x; a[1]  = (__bf16)f0.y; a[2]  = (__bf16)f0.z; a[3]  = (__bf16)f0.w;
    a[4]  = (__bf16)f1.x; a[5]  = (__bf16)f1.y; a[6]  = (__bf16)f1.z; a[7]  = (__bf16)f1.w;
    a[8]  = (__bf16)f2.x; a[9]  = (__bf16)f2.y; a[10] = (__bf16)f2.z; a[11] = (__bf16)f2.w;
    a[12] = (__bf16)f3.x; a[13] = (__bf16)f3.y; a[14] = (__bf16)f3.z; a[15] = (__bf16)f3.w;

    // ---- 4 B fragments reuse the same A fragment
    const __bf16* wr = Wt + (size_t)(k0 + half * 8) * NPAD + col0 + m;
#pragma unroll
    for (int t = 0; t < 4; ++t) {
      v16bf bf;
#pragma unroll
      for (int s = 0; s < 8; ++s) {
        bf[s]     = wr[(size_t)s * NPAD + t * 16];
        bf[8 + s] = wr[(size_t)(16 + s) * NPAD + t * 16];
      }
      acc[t] = __builtin_amdgcn_wmma_f32_16x16x32_bf16(
          /*neg_a=*/false, a, /*neg_b=*/false, bf,
          /*c_mod=*/(short)0, acc[t], /*reuse_a=*/false, /*reuse_b=*/false);
    }
  }

  // ---- Epilogue: fused bias + beta-scaled sigmoid, C/D layout writeback
#pragma unroll
  for (int t = 0; t < 4; ++t) {
    int n = col0 + t * 16 + (lane & 15);
    float bb = bpad[n];
    float bt = betapad[n];
#pragma unroll
    for (int r = 0; r < 8; ++r) {
      int mm = r + half * 8;                       // VGPR r -> row r or r+8
      float logit = acc[t][r] + bb;
      float pv = 1.0f / (1.0f + __expf(-bt * logit));
      pbuf[(size_t)(row0 + mm) * NPAD + n] = pv;
    }
  }
}

// ---------------------------------------------------------------------------
// Kernel 4: per-sample tree pass. Block = 256 threads = one sample.
// Thread t: leaf t path product, plus inner-path for node t (t<255).
// ---------------------------------------------------------------------------
__global__ void sdt_tree(const float* __restrict__ pbuf,
                         const int* __restrict__ labels,
                         const float* __restrict__ logQ,
                         const float* __restrict__ Q,
                         float* __restrict__ out,
                         float* __restrict__ loss_acc,
                         float* __restrict__ sum_inner,
                         float* __restrict__ sum_pinner) {
  const int s = blockIdx.x;
  const int t = threadIdx.x;
  __shared__ float sp[NPAD];
  __shared__ float red[NPAD];
  __shared__ int   ri[NPAD];

  sp[t] = pbuf[(size_t)s * NPAD + t];
  __syncthreads();

  // leaf-path product for leaf t (bits MSB-first pick left(1-p)/right(p))
  float path = 1.0f;
  int node = 0;
#pragma unroll
  for (int d = 0; d < 8; ++d) {
    int bit = (t >> (7 - d)) & 1;
    float pv = sp[node];
    path *= bit ? pv : (1.0f - pv);
    node = 2 * node + 1 + bit;
  }

  // loss: sum_leaf path * logQ[leaf, label]
  int label = labels[s];
  red[t] = path * logQ[(size_t)t * NOUT + label];
  __syncthreads();
  for (int ofs = 128; ofs > 0; ofs >>= 1) {
    if (t < ofs) red[t] += red[t + ofs];
    __syncthreads();
  }
  if (t == 0) atomicAdd(loss_acc, red[0]);

  // alpha accumulators: inner-path of node t
  if (t < NNODE) {
    int n1 = t + 1;                           // 1-based heap index
    int depth = 31 - __clz(n1);
    float ip = 1.0f;
    for (int d = depth - 1; d >= 0; --d) {
      int anc = n1 >> (d + 1);                // 1-based ancestor
      int bit = (n1 >> d) & 1;
      float pv = sp[anc - 1];
      ip *= bit ? pv : (1.0f - pv);
    }
    atomicAdd(&sum_inner[t], ip);
    atomicAdd(&sum_pinner[t], sp[t] * ip);
  }

  // argmax leaf_path (first max wins on ties), then gather Q row
  red[t] = path;
  ri[t] = t;
  __syncthreads();
  for (int ofs = 128; ofs > 0; ofs >>= 1) {
    if (t < ofs) {
      float ov = red[t + ofs]; int oi = ri[t + ofs];
      if (ov > red[t] || (ov == red[t] && oi < ri[t])) { red[t] = ov; ri[t] = oi; }
    }
    __syncthreads();
  }
  int best = ri[0];
  if (t < NOUT) out[1 + (size_t)s * NOUT + t] = Q[(size_t)best * NOUT + t];
}

// ---------------------------------------------------------------------------
// Kernel 5: scalars. accums[0]=loss sum; [1..256]=sum_inner; [257..512]=sum_pinner
// ---------------------------------------------------------------------------
__global__ void sdt_finalize(const float* __restrict__ accums,
                             float* __restrict__ out) {
  __shared__ float red[128];
  int t = threadIdx.x;
  float c = 0.0f;
  if (t < NPEN) {
    float a = accums[257 + t] / accums[1 + t];
    int depth = (31 - __clz(t + 1)) + 1;
    float lmb = LMBDA * exp2f(-(float)depth);
    c = lmb * 0.5f * (__logf(a) + log1pf(-a));
  }
  red[t] = c;
  __syncthreads();
  for (int ofs = 64; ofs > 0; ofs >>= 1) {
    if (t < ofs) red[t] += red[t + ofs];
    __syncthreads();
  }
  if (t == 0) {
    out[0] = -(accums[0] / (float)NBATCH);                 // neg_loss
    out[1 + (size_t)NBATCH * NOUT] = -red[0];              // C
  }
}

// ---------------------------------------------------------------------------
extern "C" void kernel_launch(void* const* d_in, const int* in_sizes, int n_in,
                              void* d_out, int out_size, void* d_ws, size_t ws_size,
                              hipStream_t stream) {
  const float* x      = (const float*)d_in[0];
  const int*   labels = (const int*)d_in[1];
  const float* W      = (const float*)d_in[2];
  const float* b      = (const float*)d_in[3];
  const float* beta   = (const float*)d_in[4];
  const float* leafp  = (const float*)d_in[5];
  float* out = (float*)d_out;

  char* ws = (char*)d_ws;
  __bf16* Wt      = (__bf16*)ws;                                   // 1 MiB
  float*  bpad    = (float*)(ws + (1u << 20));                     // 1 KiB
  float*  betapad = bpad + NPAD;                                   // 1 KiB
  float*  pbuf    = (float*)(ws + (1u << 20) + 4096);              // 8 MiB
  float*  logQ    = (float*)(ws + (1u << 20) + 4096 +
                             (size_t)NBATCH * NPAD * 4);           // 100 KiB
  float*  Q       = logQ + NPAD * NOUT;                            // 100 KiB
  float*  accums  = Q + NPAD * NOUT;                               // 513 f32

  sdt_prep<<<(DIN * NPAD) / 256, 256, 0, stream>>>(W, b, beta, Wt, bpad, betapad, accums);
  sdt_softmax<<<1, 256, 0, stream>>>(leafp, logQ, Q);
  dim3 ggrid(NBATCH / 128, NPAD / 64);
  sdt_gemm<<<ggrid, 256, 0, stream>>>(x, Wt, bpad, betapad, pbuf);
  sdt_tree<<<NBATCH, 256, 0, stream>>>(pbuf, labels, logQ, Q, out,
                                       accums, accums + 1, accums + 257);
  sdt_finalize<<<1, 128, 0, stream>>>(accums, out);
}
